// RNN_6768868458694
// MI455X (gfx1250) — matrix operand order repaired and checked by reading, not compiled, not run on hardware
//
#include <hip/hip_runtime.h>
#include <hip/hip_bf16.h>
#include <math.h>

// Problem constants (match reference)
#define BB   2048
#define TT   2048
#define DIN  6
#define HH   64
#define DOUT 6

// LDS pitches in bf16 elements (multiples of 8 -> every fragment load is 16B aligned)
#define PITCH1 104   // for K=96-padded concat input [x|h] and W1^T
#define PITCH2 72    // for K=64 hidden activations and W2^T / W3^T

#define WAVES    2
#define NTHREADS (WAVES * 32)

typedef __attribute__((ext_vector_type(16))) __bf16 v16bf;
typedef __attribute__((ext_vector_type(8)))  __bf16 v8bf;
typedef __attribute__((ext_vector_type(8)))  float  v8f;

// Per-wave ordering fence: DS ops from one wave are processed in-order by the
// LDS unit (CDNA5 ISA ch.5/11), so cross-lane store->load within a wave only
// needs the compiler not to reorder the memory ops.
__device__ __forceinline__ void wsync() {
  asm volatile("" ::: "memory");
  __builtin_amdgcn_wave_barrier();
}

__device__ __forceinline__ float fast_tanh(float v) {
#if __has_builtin(__builtin_amdgcn_tanhf)
  return __builtin_amdgcn_tanhf(v);   // gfx1250 v_tanh_f32
#else
  return tanhf(v);
#endif
}

__device__ __forceinline__ v8f vtanh(v8f c) {
  v8f r;
#pragma unroll
  for (int i = 0; i < 8; ++i) r[i] = fast_tanh(c[i]);
  return r;
}

__device__ __forceinline__ v8f vbcast(float v) {
  v8f r;
#pragma unroll
  for (int i = 0; i < 8; ++i) r[i] = v;
  return r;
}

__device__ __forceinline__ v8f wmma_bf16(v16bf a, v16bf b, v8f c) {
  return __builtin_amdgcn_wmma_f32_16x16x32_bf16(
      /*neg_a=*/false, a, /*neg_b=*/false, b,
      /*c_mod=*/(short)0, c, /*reuse_a=*/false, /*reuse_b=*/false);
}

// A fragment (16x32 bf16, row-major tile in LDS). ISA layout:
// lane L: M=L&15; VGPR0-3 hold K = kt*32 + (L<16?0:8) .. +7 ; VGPR4-7 = +16.
__device__ __forceinline__ v16bf load_A(const __bf16* base, int pitch, int kt, int lane) {
  int m  = lane & 15;
  int kb = kt * 32 + ((lane & 16) ? 8 : 0);
  const __bf16* p = base + m * pitch + kb;
  v8bf lo = *(const v8bf*)(p);
  v8bf hi = *(const v8bf*)(p + 16);
  v16bf a;
#pragma unroll
  for (int i = 0; i < 8; ++i) { a[i] = lo[i]; a[i + 8] = hi[i]; }
  return a;
}

// B fragment (32x16 bf16) from transposed weights WT[n][k]. ISA layout:
// lane L: N=L&15; VGPR0-3 hold K = kt*32 + (L<16?0:8) .. +7 ; VGPR4-7 = +16.
__device__ __forceinline__ v16bf load_B(const __bf16* wt, int pitch, int nt, int kt, int lane) {
  int n  = (nt << 4) + (lane & 15);
  int kb = kt * 32 + ((lane & 16) ? 8 : 0);
  const __bf16* p = wt + n * pitch + kb;
  v8bf lo = *(const v8bf*)(p);
  v8bf hi = *(const v8bf*)(p + 16);
  v16bf b;
#pragma unroll
  for (int i = 0; i < 8; ++i) { b[i] = lo[i]; b[i + 8] = hi[i]; }
  return b;
}

// C/D fragment (16x16 f32) -> bf16 row-major LDS tile.
// VGPR r: M = r + (lane<16?0:8), N = lane&15.
__device__ __forceinline__ void store_C(__bf16* base, int pitch, int col0, v8f c, int lane) {
  int n  = lane & 15;
  int m0 = (lane & 16) ? 8 : 0;
#pragma unroll
  for (int r = 0; r < 8; ++r)
    base[(m0 + r) * pitch + col0 + n] = (__bf16)c[r];
}

// Convert fp32 weights [K][N] row-major -> bf16 transposed WT[n][k], zero-padded.
__device__ __forceinline__ void fill_wt(__bf16* dst, const float* src,
                                        int K, int N, int pitch, int nrows, int tid) {
  for (int i = tid; i < nrows * pitch; i += NTHREADS) {
    int n = i / pitch;
    int k = i - n * pitch;
    float v = (k < K && n < N) ? src[k * N + n] : 0.0f;
    dst[i] = (__bf16)v;
  }
}

__global__ __launch_bounds__(NTHREADS)
void rnn_wmma_kernel(const float* __restrict__ x,
                     const float* __restrict__ fW1, const float* __restrict__ fb1,
                     const float* __restrict__ fW2, const float* __restrict__ fb2,
                     const float* __restrict__ gW1, const float* __restrict__ gb1,
                     const float* __restrict__ gW2, const float* __restrict__ gb2,
                     const float* __restrict__ gW3, const float* __restrict__ gb3,
                     float* __restrict__ out) {
  // Weights (shared by both waves) + per-wave staging tiles, all bf16.
  __shared__ alignas(16) __bf16 sFW1[HH * PITCH1];
  __shared__ alignas(16) __bf16 sFW2[HH * PITCH2];
  __shared__ alignas(16) __bf16 sGW1[HH * PITCH1];
  __shared__ alignas(16) __bf16 sGW2[HH * PITCH2];
  __shared__ alignas(16) __bf16 sGW3[16 * PITCH2];
  __shared__ alignas(16) __bf16 sXH [WAVES * 16 * PITCH1];   // [x(6) | h(64) | 0-pad]
  __shared__ alignas(16) __bf16 sMID[WAVES * 16 * PITCH2];   // inter-layer activations

  const int tid = threadIdx.x;

  // One-time: weights -> bf16 transposed LDS; zero staging (pad cols must be 0,
  // and h columns start at h0 = 0).
  fill_wt(sFW1, fW1, DIN + HH, HH, PITCH1, HH, tid);
  fill_wt(sFW2, fW2, HH,       HH, PITCH2, HH, tid);
  fill_wt(sGW1, gW1, DIN + HH, HH, PITCH1, HH, tid);
  fill_wt(sGW2, gW2, HH,       HH, PITCH2, HH, tid);
  fill_wt(sGW3, gW3, HH,     DOUT, PITCH2, 16, tid);
  for (int i = tid; i < WAVES * 16 * PITCH1; i += NTHREADS) sXH[i]  = (__bf16)0.0f;
  for (int i = tid; i < WAVES * 16 * PITCH2; i += NTHREADS) sMID[i] = (__bf16)0.0f;
  __syncthreads();

  const int wave = tid >> 5;
  const int lane = tid & 31;
  const int n    = lane & 15;
  const int m0   = (lane & 16) ? 8 : 0;
  const int r0   = (blockIdx.x * WAVES + wave) << 4;   // batch row base (16 rows/wave)

  __bf16* xh  = sXH  + wave * 16 * PITCH1;
  __bf16* mid = sMID + wave * 16 * PITCH2;

  // Biases live in registers, splatted per N-tile via lane column index.
  float bf1[4], bf2[4], bg1[4], bg2[4];
#pragma unroll
  for (int j = 0; j < 4; ++j) {
    bf1[j] = fb1[j * 16 + n];
    bf2[j] = fb2[j * 16 + n];
    bg1[j] = gb1[j * 16 + n];
    bg2[j] = gb2[j * 16 + n];
  }
  const float bg3 = (n < DOUT) ? gb3[n] : 0.0f;

  // Recurrent state h: 4 C-fragments (16x64 f32), kept in registers.
  v8f h[4];
#pragma unroll
  for (int j = 0; j < 4; ++j) h[j] = vbcast(0.0f);

  // x streaming: 32 lanes * 3 floats = one 16x6 tile per step. Double-buffered.
  const int xrow = lane >> 1;            // (lane*3)/6
  const int xcol = (lane & 1) * 3;       // (lane*3)%6
  const float* xbase = x + (size_t)(r0 + xrow) * (TT * DIN) + xcol;
  float xr0 = xbase[0], xr1 = xbase[1], xr2 = xbase[2];

  for (int t = 0; t < TT; ++t) {
    // --- stage x_t into concat buffer; prefetch x_{t+1} ---
    {
      __bf16* xp = xh + xrow * PITCH1 + xcol;
      xp[0] = (__bf16)xr0; xp[1] = (__bf16)xr1; xp[2] = (__bf16)xr2;
      int tn = (t + 1 < TT) ? (t + 1) : t;
      const float* p = xbase + (size_t)tn * DIN;
      xr0 = p[0]; xr1 = p[1]; xr2 = p[2];
    }
    // h_{t-1} columns already in xh (written at end of previous step / init).
    wsync();

    // --- f layer 1: t1 = tanh([x|h] @ fW1 + fb1), K=96(pad), N=64 ---
    v16bf a0 = load_A(xh, PITCH1, 0, lane);
    v16bf a1 = load_A(xh, PITCH1, 1, lane);
    v16bf a2 = load_A(xh, PITCH1, 2, lane);
#pragma unroll
    for (int j = 0; j < 4; ++j) {
      v8f acc = vbcast(bf1[j]);
      acc = wmma_bf16(a0, load_B(sFW1, PITCH1, j, 0, lane), acc);
      acc = wmma_bf16(a1, load_B(sFW1, PITCH1, j, 1, lane), acc);
      acc = wmma_bf16(a2, load_B(sFW1, PITCH1, j, 2, lane), acc);
      store_C(mid, PITCH2, j * 16, vtanh(acc), lane);
    }
    wsync();

    // --- f layer 2 + residual: h += t1 @ fW2 + fb2, K=64 ---
    {
      v16bf q0 = load_A(mid, PITCH2, 0, lane);
      v16bf q1 = load_A(mid, PITCH2, 1, lane);
#pragma unroll
      for (int j = 0; j < 4; ++j) {
        v8f acc = vbcast(bf2[j]);
        acc = wmma_bf16(q0, load_B(sFW2, PITCH2, j, 0, lane), acc);
        acc = wmma_bf16(q1, load_B(sFW2, PITCH2, j, 1, lane), acc);
        h[j] = h[j] + acc;
        store_C(xh, PITCH1, DIN + j * 16, h[j], lane);   // h_t -> concat buffer
      }
    }
    wsync();

    // --- g layer 1: u = tanh([x|h_t] @ gW1 + gb1) ---
    a0 = load_A(xh, PITCH1, 0, lane);
    a1 = load_A(xh, PITCH1, 1, lane);
    a2 = load_A(xh, PITCH1, 2, lane);
#pragma unroll
    for (int j = 0; j < 4; ++j) {
      v8f acc = vbcast(bg1[j]);
      acc = wmma_bf16(a0, load_B(sGW1, PITCH1, j, 0, lane), acc);
      acc = wmma_bf16(a1, load_B(sGW1, PITCH1, j, 1, lane), acc);
      acc = wmma_bf16(a2, load_B(sGW1, PITCH1, j, 2, lane), acc);
      store_C(mid, PITCH2, j * 16, vtanh(acc), lane);
    }
    wsync();

    // --- g layer 2: v = tanh(u @ gW2 + gb2)  (reads mid, then overwrites it;
    //     per-wave in-order DS + data deps keep the RAW/WAR legal) ---
    {
      v16bf q0 = load_A(mid, PITCH2, 0, lane);
      v16bf q1 = load_A(mid, PITCH2, 1, lane);
#pragma unroll
      for (int j = 0; j < 4; ++j) {
        v8f acc = vbcast(bg2[j]);
        acc = wmma_bf16(q0, load_B(sGW2, PITCH2, j, 0, lane), acc);
        acc = wmma_bf16(q1, load_B(sGW2, PITCH2, j, 1, lane), acc);
        store_C(mid, PITCH2, j * 16, vtanh(acc), lane);
      }
    }
    wsync();

    // --- g layer 3: y = v @ gW3 + gb3, N=6 (single N-tile) ---
    {
      v16bf q0 = load_A(mid, PITCH2, 0, lane);
      v16bf q1 = load_A(mid, PITCH2, 1, lane);
      v8f y = vbcast(bg3);
      y = wmma_bf16(q0, load_B(sGW3, PITCH2, 0, 0, lane), y);
      y = wmma_bf16(q1, load_B(sGW3, PITCH2, 0, 1, lane), y);
      if (n < DOUT) {
#pragma unroll
        for (int r = 0; r < 8; ++r)
          out[((size_t)(r0 + m0 + r) * TT + t) * DOUT + n] = y[r];
      }
    }
    wsync();
  }
}

extern "C" void kernel_launch(void* const* d_in, const int* in_sizes, int n_in,
                              void* d_out, int out_size, void* d_ws, size_t ws_size,
                              hipStream_t stream) {
  (void)in_sizes; (void)n_in; (void)out_size; (void)d_ws; (void)ws_size;
  const float* x   = (const float*)d_in[0];
  const float* fW1 = (const float*)d_in[1];
  const float* fb1 = (const float*)d_in[2];
  const float* fW2 = (const float*)d_in[3];
  const float* fb2 = (const float*)d_in[4];
  const float* gW1 = (const float*)d_in[5];
  const float* gb1 = (const float*)d_in[6];
  const float* gW2 = (const float*)d_in[7];
  const float* gb2 = (const float*)d_in[8];
  const float* gW3 = (const float*)d_in[9];
  const float* gb3 = (const float*)d_in[10];

  dim3 grid(BB / (16 * WAVES));   // 64 blocks * 2 waves = 128 batch tiles of 16
  rnn_wmma_kernel<<<grid, NTHREADS, 0, stream>>>(
      x, fW1, fb1, fW2, fb2, gW1, gb1, gW2, gb2, gW3, gb3, (float*)d_out);
}
